// Block_13812614824197
// MI455X (gfx1250) — matrix operand order repaired and checked by reading, not compiled, not run on hardware
//
#include <hip/hip_runtime.h>

typedef __attribute__((ext_vector_type(16))) __bf16 v16bf;
typedef __attribute__((ext_vector_type(8)))  __bf16 v8bf;
typedef __attribute__((ext_vector_type(8)))  float  v8f;
typedef __attribute__((ext_vector_type(4)))  int    v4i;

#define DIM     768
#define HEADS   12
#define HD      64
#define LSEQ    2048
#define BATCH   2
#define MROWS   (BATCH * LSEQ)   /* 4096 */
#define HIDDEN  3072
#define QKV_LD  (3 * DIM)        /* 2304 */
#define ATTN_SCALE 0.125f        /* 64^-0.5 */

// ---------------------------------------------------------------------------
// helpers
// ---------------------------------------------------------------------------
__device__ __forceinline__ v8f wmma_bf16(v16bf a, v16bf b, v8f c) {
  // D = A(16x32 bf16) * B(32x16 bf16) + C(16x16 f32)
  return __builtin_amdgcn_wmma_f32_16x16x32_bf16(
      /*neg_a=*/false, a, /*neg_b=*/false, b,
      /*c_mod=*/(short)0, c, /*reuse_a=*/false, /*reuse_b=*/false);
}

// Async 16-byte global -> LDS copy (ASYNCcnt path). The builtin expects
// (v4i addrspace(1)*, v4i addrspace(3)*, imm offset, imm cpol).
__device__ __forceinline__ void async_b128(const __bf16* gptr, __bf16* lptr) {
#if defined(__AMDGCN__) && __has_builtin(__builtin_amdgcn_global_load_async_to_lds_b128)
  __builtin_amdgcn_global_load_async_to_lds_b128(
      (__attribute__((address_space(1))) v4i*)gptr,
      (__attribute__((address_space(3))) v4i*)lptr,
      /*offset=*/0, /*cpol=*/0);
#else
  *(v8bf*)lptr = *(const v8bf*)gptr;
#endif
}

__device__ __forceinline__ void wait_async() {
#if defined(__AMDGCN__) && __has_builtin(__builtin_amdgcn_s_wait_asynccnt)
  __builtin_amdgcn_s_wait_asynccnt(0);
#endif
}

// A-fragment (16-bit, 16x32): lane<16 holds K in {base..base+7, 16+base..16+base+7}
// with base = 0; lane>=16 uses base = 8.  p must point at row_start + base.
__device__ __forceinline__ v16bf frag_a(const __bf16* p) {
  v8bf lo = *reinterpret_cast<const v8bf*>(p);
  v8bf hi = *reinterpret_cast<const v8bf*>(p + 16);
  v16bf r;
#pragma unroll
  for (int i = 0; i < 8; ++i) { r[i] = lo[i]; r[i + 8] = hi[i]; }
  return r;
}

// B-fragment (16-bit, 32x16): lane<16 holds K in {0..7, 16..23}, lanes 16-31
// K in {8..15, 24..31}, column = lane%16.  p points at BT[col] + half*8.
__device__ __forceinline__ v16bf frag_b(const __bf16* p) {
  v8bf lo = *reinterpret_cast<const v8bf*>(p);
  v8bf hi = *reinterpret_cast<const v8bf*>(p + 16);
  v16bf r;
#pragma unroll
  for (int i = 0; i < 8; ++i) { r[i] = lo[i]; r[i + 8] = hi[i]; }
  return r;
}

__device__ __forceinline__ v8f vzero8() {
  v8f z = {0.f, 0.f, 0.f, 0.f, 0.f, 0.f, 0.f, 0.f};
  return z;
}

// ---------------------------------------------------------------------------
// weight convert + transpose: f32 [K][N] -> bf16 [N][K]
// ---------------------------------------------------------------------------
__global__ __launch_bounds__(256) void wt_kernel(const float* __restrict__ w,
                                                 __bf16* __restrict__ wt,
                                                 int K, int N) {
  size_t i = (size_t)blockIdx.x * 256 + threadIdx.x;
  if (i >= (size_t)K * N) return;
  int n = (int)(i / K);
  int k = (int)(i % K);
  wt[i] = (__bf16)w[(size_t)k * N + n];
}

// ---------------------------------------------------------------------------
// layernorm(x) * w -> bf16 ; one row (DIM floats) per 256-thread block
// ---------------------------------------------------------------------------
__global__ __launch_bounds__(256) void ln_bf16_kernel(
    const float* __restrict__ x, const float* __restrict__ w,
    __bf16* __restrict__ out) {
  const int row = blockIdx.x;
  const float* xr = x + (size_t)row * DIM;
  float s = 0.f, s2 = 0.f;
  for (int c = threadIdx.x; c < DIM; c += 256) {
    float v = xr[c];
    s += v;
    s2 += v * v;
  }
#pragma unroll
  for (int off = 16; off > 0; off >>= 1) {
    s += __shfl_xor(s, off, 32);
    s2 += __shfl_xor(s2, off, 32);
  }
  __shared__ float ss[8], ss2[8];
  const int wid = threadIdx.x >> 5, lane = threadIdx.x & 31;
  if (lane == 0) { ss[wid] = s; ss2[wid] = s2; }
  __syncthreads();
  if (threadIdx.x == 0) {
    float a = 0.f, b = 0.f;
#pragma unroll
    for (int i = 0; i < 8; ++i) { a += ss[i]; b += ss2[i]; }
    ss[0] = a;
    ss2[0] = b;
  }
  __syncthreads();
  const float mu = ss[0] * (1.f / DIM);
  const float var = ss2[0] * (1.f / DIM) - mu * mu;
  const float rinv = rsqrtf(var + 1e-6f);
  for (int c = threadIdx.x; c < DIM; c += 256)
    out[(size_t)row * DIM + c] = (__bf16)((xr[c] - mu) * rinv * w[c]);
}

// ---------------------------------------------------------------------------
// WMMA GEMM:  C[M,N] = A[M,K](bf16,row-major) * B  with B given transposed as
// Bt[N,K](bf16,row-major).  Block tile 128x128, 8 waves (2x4), wave = 64x32.
// Double-buffered LDS with async global->LDS copies pipelined against WMMA.
// MODE 0: out bf16 = C     (QKV)
// MODE 1: out f32  = C + res (proj / mlp2 residual add)
// MODE 2: out bf16 = gelu(C) (mlp1)
// ---------------------------------------------------------------------------
template <int MODE>
__global__ __launch_bounds__(256) void gemm_bf16_kernel(
    const __bf16* __restrict__ A, const __bf16* __restrict__ Bt,
    __bf16* __restrict__ outb, float* __restrict__ outf,
    const float* __restrict__ res, int N, int K) {
  __shared__ __bf16 As[2][128][40];   // 128 x 32, +8 pad (80B row stride)
  __shared__ __bf16 Bs[2][128][40];   // Bs[buf][n][k]

  const int tid = threadIdx.x;
  const int wid = tid >> 5, lane = tid & 31;
  const int half = lane >> 4, l16 = lane & 15;
  const int wm = wid >> 2, wn = wid & 3;
  const int bm = blockIdx.y, bn = blockIdx.x;

  v8f acc[4][2];
#pragma unroll
  for (int mt = 0; mt < 4; ++mt)
#pragma unroll
    for (int nt = 0; nt < 2; ++nt) acc[mt][nt] = vzero8();

  const int fr = tid >> 1;          // fill row 0..127
  const int fc = (tid & 1) * 16;    // fill col chunk 0/16
  const __bf16* ga = A + (size_t)(bm * 128 + fr) * K + fc;
  const __bf16* gb = Bt + (size_t)(bn * 128 + fr) * K + fc;

  // prologue: async-fill buffer 0 with the first K-slab
  async_b128(ga, &As[0][fr][fc]);
  async_b128(ga + 8, &As[0][fr][fc + 8]);
  async_b128(gb, &Bs[0][fr][fc]);
  async_b128(gb + 8, &Bs[0][fr][fc + 8]);
  wait_async();
  __syncthreads();

  int buf = 0;
  for (int k0 = 0; k0 < K; k0 += 32) {
    const int nk = k0 + 32;
    if (nk < K) {  // prefetch next slab into the other buffer (async, no VGPRs)
      async_b128(ga + nk, &As[buf ^ 1][fr][fc]);
      async_b128(ga + nk + 8, &As[buf ^ 1][fr][fc + 8]);
      async_b128(gb + nk, &Bs[buf ^ 1][fr][fc]);
      async_b128(gb + nk + 8, &Bs[buf ^ 1][fr][fc + 8]);
    }

    v16bf bfr[2];
#pragma unroll
    for (int nt = 0; nt < 2; ++nt)
      bfr[nt] = frag_b(&Bs[buf][wn * 32 + nt * 16 + l16][half * 8]);
#pragma unroll
    for (int mt = 0; mt < 4; ++mt) {
      v16bf afr = frag_a(&As[buf][wm * 64 + mt * 16 + l16][half * 8]);
#pragma unroll
      for (int nt = 0; nt < 2; ++nt)
        acc[mt][nt] = wmma_bf16(afr, bfr[nt], acc[mt][nt]);
    }

    wait_async();
    __syncthreads();
    buf ^= 1;
  }

  // Epilogue.  C layout: VGPR r, lanes 0-15 -> M = +r, lanes 16-31 -> M = 8+r.
#pragma unroll
  for (int mt = 0; mt < 4; ++mt)
#pragma unroll
    for (int nt = 0; nt < 2; ++nt)
#pragma unroll
      for (int r = 0; r < 8; ++r) {
        const int row = bm * 128 + wm * 64 + mt * 16 + half * 8 + r;
        const int col = bn * 128 + wn * 32 + nt * 16 + l16;
        const size_t idx = (size_t)row * N + col;
        const float v = acc[mt][nt][r];
        if (MODE == 0) {
          outb[idx] = (__bf16)v;
        } else if (MODE == 1) {
          outf[idx] = v + res[idx];
        } else {  // exact GELU
          outb[idx] = (__bf16)(0.5f * v * (1.0f + erff(v * 0.7071067811865476f)));
        }
      }
}

// ---------------------------------------------------------------------------
// Flash attention, bf16 WMMA, fp32 online softmax.
// grid = (L/128 q-tiles, B*H).  256 threads = 8 waves; each wave owns 16 q rows
// so softmax row reductions are intra-half-wave shuffles only.
// ---------------------------------------------------------------------------
#define ATTN_SMEM_BYTES ((128 * 72 + 128 * 72 + 64 * 136 + 8 * 16 * 136) * 2)

__global__ __launch_bounds__(256) void attn_kernel(
    const __bf16* __restrict__ qkv, __bf16* __restrict__ o) {
  extern __shared__ unsigned char smem_raw[];
  __bf16* Qs = (__bf16*)smem_raw;     // [128][72]  q rows, hd-contig
  __bf16* Ks = Qs + 128 * 72;         // [128][72]  key rows, hd-contig (== BT for QK^T)
  __bf16* Vs = Ks + 128 * 72;         // [64][136]  V transposed: [d][key]
  __bf16* Ps = Vs + 64 * 136;         // [8][16][136] per-wave P staging (C->A relayout)

  const int tid = threadIdx.x;
  const int wid = tid >> 5, lane = tid & 31;
  const int half = lane >> 4, l16 = lane & 15;
  const int bh = blockIdx.y;
  const int b = bh / HEADS, h = bh % HEADS;
  const int q0 = blockIdx.x * 128;
  const size_t rowbase = (size_t)b * LSEQ;
  const int qoff = h * (3 * HD);
  const int koff = qoff + HD, voff = qoff + 2 * HD;

  const int fr = tid >> 1;          // 0..127
  const int fc = (tid & 1) * 32;    // 0/32

  {  // load Q tile once (async -> LDS, waited inside first key-tile iteration)
    const __bf16* g = qkv + (rowbase + q0 + fr) * QKV_LD + qoff + fc;
#pragma unroll
    for (int j = 0; j < 4; ++j)
      async_b128(g + j * 8, &Qs[fr * 72 + fc + j * 8]);
  }

  float mprev[8], lprev[8];
#pragma unroll
  for (int r = 0; r < 8; ++r) { mprev[r] = -3.0e38f; lprev[r] = 0.f; }
  v8f oacc[4];  // 16 q x 64 d
#pragma unroll
  for (int ot = 0; ot < 4; ++ot) oacc[ot] = vzero8();

  for (int kt = 0; kt < LSEQ; kt += 128) {
    __syncthreads();
    {  // fill K rows (async) and transposed V (through VGPRs, must transpose)
      const __bf16* gk = qkv + (rowbase + kt + fr) * QKV_LD + koff + fc;
#pragma unroll
      for (int j = 0; j < 4; ++j)
        async_b128(gk + j * 8, &Ks[fr * 72 + fc + j * 8]);
      const __bf16* gv = qkv + (rowbase + kt + fr) * QKV_LD + voff + fc;
#pragma unroll
      for (int j = 0; j < 4; ++j) {
        v8bf t = *(const v8bf*)(gv + j * 8);
#pragma unroll
        for (int e = 0; e < 8; ++e)
          Vs[(fc + j * 8 + e) * 136 + fr] = t[e];
      }
    }
    wait_async();
    __syncthreads();

    // S = Q K^T  (16 q x 128 keys per wave, 8 fragments)
    v8f s[8];
#pragma unroll
    for (int nt = 0; nt < 8; ++nt) s[nt] = vzero8();
#pragma unroll
    for (int kk = 0; kk < 2; ++kk) {
      v16bf a = frag_a(&Qs[(wid * 16 + l16) * 72 + kk * 32 + half * 8]);
#pragma unroll
      for (int nt = 0; nt < 8; ++nt) {
        v16bf bb = frag_b(&Ks[(nt * 16 + l16) * 72 + kk * 32 + half * 8]);
        s[nt] = wmma_bf16(a, bb, s[nt]);
      }
    }

    // online softmax per row (row r lives in VGPR r, half-wave 'half')
#pragma unroll
    for (int r = 0; r < 8; ++r) {
      float mx = -3.0e38f;
#pragma unroll
      for (int nt = 0; nt < 8; ++nt) mx = fmaxf(mx, s[nt][r]);
#pragma unroll
      for (int off = 1; off < 16; off <<= 1)
        mx = fmaxf(mx, __shfl_xor(mx, off, 32));
      mx *= ATTN_SCALE;
      const float mnew = fmaxf(mprev[r], mx);
      const float alpha = __expf(mprev[r] - mnew);
      float rowsum = 0.f;
#pragma unroll
      for (int nt = 0; nt < 8; ++nt) {
        const float p = __expf(s[nt][r] * ATTN_SCALE - mnew);
        rowsum += p;
        Ps[(wid * 16 + half * 8 + r) * 136 + nt * 16 + l16] = (__bf16)p;
      }
#pragma unroll
      for (int off = 1; off < 16; off <<= 1)
        rowsum += __shfl_xor(rowsum, off, 32);
      lprev[r] = lprev[r] * alpha + rowsum;
      mprev[r] = mnew;
#pragma unroll
      for (int ot = 0; ot < 4; ++ot) oacc[ot][r] *= alpha;
    }

    // O += P * V   (P is per-wave in LDS; same-wave DS ops are in order)
#pragma unroll
    for (int kk = 0; kk < 4; ++kk) {
      v16bf a = frag_a(&Ps[(wid * 16 + l16) * 136 + kk * 32 + half * 8]);
#pragma unroll
      for (int ot = 0; ot < 4; ++ot) {
        v16bf bb = frag_b(&Vs[(ot * 16 + l16) * 136 + kk * 32 + half * 8]);
        oacc[ot] = wmma_bf16(a, bb, oacc[ot]);
      }
    }
  }

  // normalize and store o[b, q, h, d] (bf16, row-major [M][DIM])
#pragma unroll
  for (int ot = 0; ot < 4; ++ot)
#pragma unroll
    for (int r = 0; r < 8; ++r) {
      const int row = q0 + wid * 16 + half * 8 + r;
      const int col = ot * 16 + l16;
      const float inv = 1.f / lprev[r];
      o[(rowbase + row) * DIM + h * HD + col] = (__bf16)(oacc[ot][r] * inv);
    }
}

// ---------------------------------------------------------------------------
// host launcher
// ---------------------------------------------------------------------------
extern "C" void kernel_launch(void* const* d_in, const int* in_sizes, int n_in,
                              void* d_out, int out_size, void* d_ws,
                              size_t ws_size, hipStream_t stream) {
  (void)in_sizes; (void)n_in; (void)out_size; (void)ws_size;
  const float* x      = (const float*)d_in[0];
  // d_in[1] = mask (all-true) -> ignored
  const float* ln1_w  = (const float*)d_in[2];
  const float* ln2_w  = (const float*)d_in[3];
  const float* w_qkv  = (const float*)d_in[4];
  const float* w_proj = (const float*)d_in[5];
  const float* w_mlp1 = (const float*)d_in[6];
  const float* w_mlp2 = (const float*)d_in[7];
  float* out = (float*)d_out;

  char* ws = (char*)d_ws;
  size_t off = 0;
  auto alloc = [&](size_t bytes) -> void* {
    void* p = ws + off;
    off += (bytes + 255) & ~(size_t)255;
    return p;
  };
  __bf16* h1     = (__bf16*)alloc((size_t)MROWS * DIM * 2);
  __bf16* wqkvT  = (__bf16*)alloc((size_t)QKV_LD * DIM * 2);
  __bf16* wprojT = (__bf16*)alloc((size_t)DIM * DIM * 2);
  __bf16* wm1T   = (__bf16*)alloc((size_t)HIDDEN * DIM * 2);
  __bf16* wm2T   = (__bf16*)alloc((size_t)DIM * HIDDEN * 2);
  __bf16* qkvb   = (__bf16*)alloc((size_t)MROWS * QKV_LD * 2);
  __bf16* ob     = (__bf16*)alloc((size_t)MROWS * DIM * 2);
  float*  x1     = (float*) alloc((size_t)MROWS * DIM * 4);
  __bf16* h2     = (__bf16*)alloc((size_t)MROWS * DIM * 2);
  __bf16* gbuf   = (__bf16*)alloc((size_t)MROWS * HIDDEN * 2);

  // 1. weight convert+transpose to bf16 [N][K]
  {
    size_t t;
    t = (size_t)DIM * QKV_LD;
    wt_kernel<<<dim3((unsigned)((t + 255) / 256)), 256, 0, stream>>>(w_qkv, wqkvT, DIM, QKV_LD);
    t = (size_t)DIM * DIM;
    wt_kernel<<<dim3((unsigned)((t + 255) / 256)), 256, 0, stream>>>(w_proj, wprojT, DIM, DIM);
    t = (size_t)DIM * HIDDEN;
    wt_kernel<<<dim3((unsigned)((t + 255) / 256)), 256, 0, stream>>>(w_mlp1, wm1T, DIM, HIDDEN);
    t = (size_t)HIDDEN * DIM;
    wt_kernel<<<dim3((unsigned)((t + 255) / 256)), 256, 0, stream>>>(w_mlp2, wm2T, HIDDEN, DIM);
  }

  // 2. ln1(x) -> h1 (bf16)
  ln_bf16_kernel<<<MROWS, 256, 0, stream>>>(x, ln1_w, h1);
  // 3. qkv = h1 @ w_qkv  -> bf16
  gemm_bf16_kernel<0><<<dim3(QKV_LD / 128, MROWS / 128), 256, 0, stream>>>(
      h1, wqkvT, qkvb, nullptr, nullptr, QKV_LD, DIM);
  // 4. flash attention -> ob (bf16)
  attn_kernel<<<dim3(LSEQ / 128, BATCH * HEADS), 256, ATTN_SMEM_BYTES, stream>>>(
      qkvb, ob);
  // 5. x1 = x + ob @ w_proj (f32)
  gemm_bf16_kernel<1><<<dim3(DIM / 128, MROWS / 128), 256, 0, stream>>>(
      ob, wprojT, nullptr, x1, x, DIM, DIM);
  // 6. ln2(x1) -> h2 (bf16)
  ln_bf16_kernel<<<MROWS, 256, 0, stream>>>(x1, ln2_w, h2);
  // 7. g = gelu(h2 @ w_mlp1) -> bf16
  gemm_bf16_kernel<2><<<dim3(HIDDEN / 128, MROWS / 128), 256, 0, stream>>>(
      h2, wm1T, gbuf, nullptr, nullptr, HIDDEN, DIM);
  // 8. out = x1 + g @ w_mlp2 (f32)
  gemm_bf16_kernel<1><<<dim3(DIM / 128, MROWS / 128), 256, 0, stream>>>(
      gbuf, wm2T, nullptr, out, x1, DIM, HIDDEN);
}
